// GCN64Model_16071767622240
// MI455X (gfx1250) — matrix operand order repaired and checked by reading, not compiled, not run on hardware
//
#include <hip/hip_runtime.h>

// ---------------------------------------------------------------------------
// CDNA5 (gfx1250) implementation of the dual-GCN model.
// Dense layers: bf16 WMMA (v_wmma_f32_16x16x32_bf16), 8 waves/block,
// 128-row M tiles. A staged fp32->bf16 via float4 loads; weights staged with
// global_load_async_to_lds_b128 (ASYNCcnt). Graph aggregation: device-built
// CSR + wave-per-node gather (no fp atomics on the output).
// ---------------------------------------------------------------------------

typedef __bf16 bhalf_t;
typedef __attribute__((ext_vector_type(4)))  __bf16 v4bf;
typedef __attribute__((ext_vector_type(8)))  __bf16 v8bf;
typedef __attribute__((ext_vector_type(16))) __bf16 v16bf;
typedef __attribute__((ext_vector_type(8)))  float  v8f;

__device__ __forceinline__ bhalf_t f2bf(float f) {
  unsigned u = __builtin_bit_cast(unsigned, f);
  unsigned r = u + 0x7FFFu + ((u >> 16) & 1u);   // round-to-nearest-even
  unsigned short h = (unsigned short)(r >> 16);
  return __builtin_bit_cast(bhalf_t, h);
}

// ---------------------------------------------------------------------------
// Weight prep: W [K x Nout] fp32 row-major  ->  WT [Nout x K] bf16 (K-contig)
// ---------------------------------------------------------------------------
__global__ void k_wt(const float* __restrict__ W, bhalf_t* __restrict__ WT,
                     int K, int Nout) {
  int i = blockIdx.x * 256 + threadIdx.x;
  if (i < K * Nout) {
    int k = i / Nout, n = i - k * Nout;
    WT[n * K + k] = f2bf(W[i]);
  }
}

// ---------------------------------------------------------------------------
// WMMA GEMM:  Out[M x Nout] = concat(A0[:,0:K0], A1[:,0:K1]) @ W  (+epilogue)
//   KTOT = K0+K1 (compile time), NT = Nout/16 (compile time)
//   Block: 256 threads (8 waves), tile 128 x Nout. Wave w owns rows w*16..+15.
// Fragment layouts per CDNA5 ISA 7.12.2:
//   A 16x32 bf16 : lane = {half=l/16, r=l%16}; elems 0-7 = K[koff..koff+7],
//                  elems 8-15 = K[koff+16..koff+23], koff = kstep + half*8
//   B 32x16 bf16 : col = l%16; elems 0-7 = K[koff..koff+7], 8-15 = K[koff+16..koff+23] (same as A)
//   C/D 16x16 f32: col = l%16; vgpr r -> row = half*8 + r
// ---------------------------------------------------------------------------
template <int KTOT, int NT>
__global__ __launch_bounds__(256) void gemm_kernel(
    const float* __restrict__ A0, int lda0, int K0,
    const float* __restrict__ A1, int lda1,
    const bhalf_t* __restrict__ WT,
    const float* __restrict__ bias,
    const float* __restrict__ bn_g, const float* __restrict__ bn_b,
    const float* __restrict__ bn_m, const float* __restrict__ bn_v,
    int relu, float* __restrict__ Out, int ldo, int M) {
  constexpr int LDK = KTOT + 8;                 // pad: breaks 256B bank period
  extern __shared__ char smem_raw[];
  bhalf_t* sA = (bhalf_t*)smem_raw;             // [128][LDK]
  bhalf_t* sW = sA + 128 * LDK;                 // [NT*16][LDK]

  const int tid  = threadIdx.x;
  const int wave = tid >> 5;
  const int lane = tid & 31;
  const int half = lane >> 4;
  const int l16  = lane & 15;
  const int rowBlk = blockIdx.x * 128;

  // ---- stage weights via async global->LDS DMA (16B chunks, pad-aware) ----
  {
    constexpr int WBYTES  = NT * 16 * KTOT * 2;   // contiguous source bytes
    constexpr int ROWB    = KTOT * 2;             // bytes per weight row
    constexpr int LDSROWB = LDK * 2;              // bytes per padded LDS row
    const unsigned sWbase = (unsigned)(size_t)sW; // low 32 bits = LDS offset
#pragma unroll
    for (int bo = 0; bo < WBYTES; bo += 256 * 16) {
      int b = bo + tid * 16;
      int n = b / ROWB;
      int rem = b - n * ROWB;
      unsigned ldsAddr = sWbase + (unsigned)(n * LDSROWB + rem);
      unsigned long long gaddr = (unsigned long long)(const char*)WT + (unsigned)b;
      asm volatile("global_load_async_to_lds_b128 %0, %1, off"
                   :: "v"(ldsAddr), "v"(gaddr) : "memory");
    }
  }

  // ---- stage A tile fp32 -> bf16 (float4 loads, b64 LDS stores) ----------
  {
    constexpr int K4 = KTOT / 4;
    for (int idx = tid; idx < 128 * K4; idx += 256) {
      int r = idx / K4, k4 = (idx - r * K4) * 4;
      int row = rowBlk + r;
      float4 v = make_float4(0.f, 0.f, 0.f, 0.f);
      if (row < M) {
        const float* src = (k4 < K0) ? (A0 + (size_t)row * lda0 + k4)
                                     : (A1 + (size_t)row * lda1 + (k4 - K0));
        v = *(const float4*)src;
      }
      v4bf p = {f2bf(v.x), f2bf(v.y), f2bf(v.z), f2bf(v.w)};
      *(v4bf*)(sA + r * LDK + k4) = p;
    }
  }
  asm volatile("s_wait_asynccnt 0x0" ::: "memory");
  __syncthreads();

  v8f acc[NT];
  {
    v8f zf = {0.f, 0.f, 0.f, 0.f, 0.f, 0.f, 0.f, 0.f};
#pragma unroll
    for (int t = 0; t < NT; ++t) acc[t] = zf;
  }

  const bhalf_t* aRow = sA + (wave * 16 + l16) * LDK;
#pragma unroll
  for (int k = 0; k < KTOT; k += 32) {
    const int koff = k + half * 8;
    v8bf alo = *(const v8bf*)(aRow + koff);
    v8bf ahi = *(const v8bf*)(aRow + koff + 16);
    v16bf afrag = __builtin_shufflevector(alo, ahi, 0, 1, 2, 3, 4, 5, 6, 7, 8,
                                          9, 10, 11, 12, 13, 14, 15);
    const int kb = k + half * 8;
    // preload all B fragments for this k-step, then issue WMMAs back-to-back
    v16bf bfr[NT];
#pragma unroll
    for (int t = 0; t < NT; ++t) {
      const bhalf_t* bp = sW + (t * 16 + l16) * LDK + kb;
      v8bf blo = *(const v8bf*)(bp);
      v8bf bhi = *(const v8bf*)(bp + 16);
      bfr[t] = __builtin_shufflevector(blo, bhi, 0, 1, 2, 3, 4, 5, 6, 7, 8, 9,
                                       10, 11, 12, 13, 14, 15);
    }
    // hard scheduling fence: keep the ds_load burst before the WMMA burst so
    // the backend cannot re-sink loads (one s_wait_dscnt, NT WMMAs back-to-back)
    __builtin_amdgcn_sched_barrier(0);
#pragma unroll
    for (int t = 0; t < NT; ++t) {
      acc[t] = __builtin_amdgcn_wmma_f32_16x16x32_bf16(
          false, afrag, false, bfr[t], (short)0, acc[t], false, false);
    }
    __builtin_amdgcn_sched_barrier(0);
  }

  // epilogue: bias + folded BatchNorm + ReLU
#pragma unroll
  for (int t = 0; t < NT; ++t) {
    int col = t * 16 + l16;
    float scale = 1.0f, shift = 0.0f;
    if (bias) shift = bias[col];
    if (bn_g) {
      float s = bn_g[col] * rsqrtf(bn_v[col] + 1e-5f);
      shift = shift * s + bn_b[col] - bn_m[col] * s;
      scale = s;
    }
#pragma unroll
    for (int r = 0; r < 8; ++r) {
      int row = rowBlk + wave * 16 + half * 8 + r;
      if (row < M) {
        float y = acc[t][r] * scale + shift;
        if (relu) y = fmaxf(y, 0.0f);
        Out[(size_t)row * ldo + col] = y;
      }
    }
  }
}

template <int KTOT, int NT>
static inline void rungemm(const float* A0, int lda0, int K0, const float* A1,
                           int lda1, const bhalf_t* WT, const float* bias,
                           const float* g, const float* b, const float* m,
                           const float* v, int relu, float* Out, int ldo,
                           int M, hipStream_t s) {
  size_t sh = (size_t)(128 + NT * 16) * (KTOT + 8) * sizeof(bhalf_t);
  hipFuncSetAttribute(reinterpret_cast<const void*>(gemm_kernel<KTOT, NT>),
                      hipFuncAttributeMaxDynamicSharedMemorySize, (int)sh);
  gemm_kernel<KTOT, NT><<<(M + 127) / 128, 256, sh, s>>>(
      A0, lda0, K0, A1, lda1, WT, bias, g, b, m, v, relu, Out, ldo, M);
}

// ---------------------------------------------------------------------------
// Graph build: histogram -> exclusive scan -> atomic-cursor CSR fill
// ---------------------------------------------------------------------------
__global__ void k_count(const int* __restrict__ dst, unsigned* __restrict__ cnt,
                        int E) {
  int i = blockIdx.x * blockDim.x + threadIdx.x;
  if (i < E) atomicAdd(&cnt[dst[i]], 1u);
}

// exclusive scan, 1024/block; sums (optional) gets per-block totals.
__global__ void k_scan1(const unsigned* __restrict__ in,
                        unsigned* __restrict__ out,
                        unsigned* __restrict__ sums, int n) {
  __shared__ unsigned sh[1024];
  int i = blockIdx.x * 1024 + threadIdx.x;
  unsigned x = (i < n) ? in[i] : 0u;
  sh[threadIdx.x] = x;
  __syncthreads();
  for (int off = 1; off < 1024; off <<= 1) {
    unsigned v = (threadIdx.x >= off) ? sh[threadIdx.x - off] : 0u;
    __syncthreads();
    sh[threadIdx.x] += v;
    __syncthreads();
  }
  if (i < n) out[i] = sh[threadIdx.x] - x;  // exclusive
  if (sums && threadIdx.x == 1023) sums[blockIdx.x] = sh[1023];
}

__global__ void k_scan_add(unsigned* __restrict__ out,
                           const unsigned* __restrict__ sums, int n) {
  int i = blockIdx.x * 1024 + threadIdx.x;
  if (i < n) out[i] += sums[blockIdx.x];
}

__global__ void k_dis(const unsigned* __restrict__ cnt, float* __restrict__ dis,
                      int n) {
  int i = blockIdx.x * blockDim.x + threadIdx.x;
  if (i < n) dis[i] = rsqrtf((float)cnt[i] + 1.0f);
}

__global__ void k_fill(const int* __restrict__ src, const int* __restrict__ dst,
                       const unsigned* __restrict__ rowptr,
                       unsigned* __restrict__ cursor,
                       unsigned* __restrict__ colidx, int E) {
  int i = blockIdx.x * blockDim.x + threadIdx.x;
  if (i < E) {
    int d = dst[i];
    unsigned p = rowptr[d] + atomicAdd(&cursor[d], 1u);
    colidx[p] = (unsigned)src[i];
  }
}

// wave-per-node aggregation: out[d] = relu( sum_e dis[s]dis[d] h[s]
//                                           + h[d]/deg[d] + bias )
__global__ __launch_bounds__(256) void k_aggregate(
    const unsigned* __restrict__ rowptr, const unsigned* __restrict__ cnt,
    const unsigned* __restrict__ colidx, const float* __restrict__ dis,
    const float* __restrict__ hlin, const float* __restrict__ bias,
    float* __restrict__ out, int ldo, int colOff, int n) {
  int d = blockIdx.x * 8 + (threadIdx.x >> 5);
  if (d >= n) return;
  int lane = threadIdx.x & 31;
  int c = lane * 2;  // 64 channels, 2 per lane
  unsigned beg = rowptr[d], num = cnt[d];
  float disd = dis[d];
  float a0 = 0.f, a1 = 0.f;
  for (unsigned e = 0; e < num; ++e) {
    // src index is wave-uniform: force scalarization of dis/row lookups
    unsigned s = (unsigned)__builtin_amdgcn_readfirstlane(
        (int)colidx[beg + e]);
    if (e + 8u < num) __builtin_prefetch(&colidx[beg + e + 8u], 0, 0);
    float w = dis[s] * disd;
    float2 hv = *(const float2*)(hlin + (size_t)s * 64 + c);
    a0 = fmaf(w, hv.x, a0);
    a1 = fmaf(w, hv.y, a1);
  }
  float sw = disd * disd;  // = 1/deg (self-loop term)
  float2 hd = *(const float2*)(hlin + (size_t)d * 64 + c);
  float y0 = fmaf(sw, hd.x, a0) + bias[c];
  float y1 = fmaf(sw, hd.y, a1) + bias[c + 1];
  out[(size_t)d * ldo + colOff + c] = fmaxf(y0, 0.f);
  out[(size_t)d * ldo + colOff + c + 1] = fmaxf(y1, 0.f);
}

// ---------------------------------------------------------------------------
extern "C" void kernel_launch(void* const* d_in, const int* in_sizes, int n_in,
                              void* d_out, int out_size, void* d_ws,
                              size_t ws_size, hipStream_t stream) {
  const int N = in_sizes[0] / 256;   // 100000
  const int E = in_sizes[2] / 2;     // 3200000

  const float* ctx     = (const float*)d_in[0];
  const float* vis     = (const float*)d_in[1];
  const int*   sp_ei   = (const int*)d_in[2];
  const int*   tr_ei   = (const int*)d_in[3];
  const float* ctx_w1  = (const float*)d_in[4];
  const float* ctx_b1  = (const float*)d_in[5];
  const float* bn1_g   = (const float*)d_in[6];
  const float* bn1_b   = (const float*)d_in[7];
  const float* bn1_m   = (const float*)d_in[8];
  const float* bn1_v   = (const float*)d_in[9];
  const float* ctx_w2  = (const float*)d_in[10];
  const float* ctx_b2  = (const float*)d_in[11];
  const float* bn2_g   = (const float*)d_in[12];
  const float* bn2_b   = (const float*)d_in[13];
  const float* bn2_m   = (const float*)d_in[14];
  const float* bn2_v   = (const float*)d_in[15];
  const float* tgt_w1  = (const float*)d_in[16];
  const float* tgt_b1  = (const float*)d_in[17];
  const float* tgt_w2  = (const float*)d_in[18];
  const float* tgt_b2  = (const float*)d_in[19];
  const float* spa_w   = (const float*)d_in[20];
  const float* spa_b   = (const float*)d_in[21];
  const float* tra_w   = (const float*)d_in[22];
  const float* tra_b   = (const float*)d_in[23];
  const float* fuse_w  = (const float*)d_in[24];
  const float* fuse_b  = (const float*)d_in[25];
  const float* head_w1 = (const float*)d_in[26];
  const float* head_b1 = (const float*)d_in[27];
  const float* bn3_g   = (const float*)d_in[28];
  const float* bn3_b   = (const float*)d_in[29];
  const float* bn3_m   = (const float*)d_in[30];
  const float* bn3_v   = (const float*)d_in[31];
  const float* head_w2 = (const float*)d_in[32];
  const float* head_b2 = (const float*)d_in[33];
  const float* head_w3 = (const float*)d_in[34];
  const float* head_b3 = (const float*)d_in[35];

  // ---- workspace layout -------------------------------------------------
  char* base = (char*)d_ws;
  size_t off = 0;
  auto alloc = [&](size_t bytes) -> void* {
    void* p = base + off;
    off = (off + bytes + 255) & ~(size_t)255;
    return p;
  };
  bhalf_t* w1t  = (bhalf_t*)alloc(256 * 128 * 2);
  bhalf_t* w2t  = (bhalf_t*)alloc(128 * 128 * 2);
  bhalf_t* wt1t = (bhalf_t*)alloc(64 * 64 * 2);
  bhalf_t* wt2t = (bhalf_t*)alloc(64 * 64 * 2);
  bhalf_t* wspt = (bhalf_t*)alloc(192 * 64 * 2);
  bhalf_t* wtrt = (bhalf_t*)alloc(192 * 64 * 2);
  bhalf_t* wfut = (bhalf_t*)alloc(128 * 64 * 2);
  bhalf_t* wh1t = (bhalf_t*)alloc(256 * 128 * 2);
  bhalf_t* wh2t = (bhalf_t*)alloc(128 * 64 * 2);
  bhalf_t* wh3t = (bhalf_t*)alloc(64 * 64 * 2);
  float* fus  = (float*)alloc((size_t)N * 192 * 4);  // concat(c, t)
  float* tmpA = (float*)alloc((size_t)N * 128 * 4);  // ctx hidden / head z1
  float* hcat = (float*)alloc((size_t)N * 128 * 4);  // concat(hs, ht)
  float* hsl  = (float*)alloc((size_t)N * 64 * 4);   // fus @ spa_w
  float* htl  = (float*)alloc((size_t)N * 64 * 4);   // fus @ tra_w
  float* z2   = (float*)alloc((size_t)N * 64 * 4);   // small hidden
  unsigned* cnt_sp = (unsigned*)alloc((size_t)N * 4);
  unsigned* cnt_tr = (unsigned*)alloc((size_t)N * 4);
  unsigned* cur_sp = (unsigned*)alloc((size_t)N * 4);
  unsigned* cur_tr = (unsigned*)alloc((size_t)N * 4);
  unsigned* rp_sp  = (unsigned*)alloc((size_t)N * 4);
  unsigned* rp_tr  = (unsigned*)alloc((size_t)N * 4);
  float* dis_sp = (float*)alloc((size_t)N * 4);
  float* dis_tr = (float*)alloc((size_t)N * 4);
  unsigned* col_sp  = (unsigned*)alloc((size_t)E * 4);
  unsigned* col_tr  = (unsigned*)alloc((size_t)E * 4);
  unsigned* sums_sp = (unsigned*)alloc(1024 * 4);
  unsigned* sums_tr = (unsigned*)alloc(1024 * 4);

  float* y_out = (float*)d_out;                       // [N,64] final
  float* h_out = (float*)d_out + (size_t)N * 64;      // [N,64] second output

  // ---- weight conversion (bf16, transposed) -----------------------------
  auto wt = [&](const float* W, bhalf_t* WT, int K, int Nout) {
    int tot = K * Nout;
    k_wt<<<(tot + 255) / 256, 256, 0, stream>>>(W, WT, K, Nout);
  };
  wt(ctx_w1, w1t, 256, 128);
  wt(ctx_w2, w2t, 128, 128);
  wt(tgt_w1, wt1t, 64, 64);
  wt(tgt_w2, wt2t, 64, 64);
  wt(spa_w, wspt, 192, 64);
  wt(tra_w, wtrt, 192, 64);
  wt(fuse_w, wfut, 128, 64);
  wt(head_w1, wh1t, 256, 128);
  wt(head_w2, wh2t, 128, 64);
  wt(head_w3, wh3t, 64, 64);

  hipMemsetAsync(cnt_sp, 0, (size_t)N * 4, stream);
  hipMemsetAsync(cnt_tr, 0, (size_t)N * 4, stream);
  hipMemsetAsync(cur_sp, 0, (size_t)N * 4, stream);
  hipMemsetAsync(cur_tr, 0, (size_t)N * 4, stream);

  // ---- towers -----------------------------------------------------------
  rungemm<256, 8>(ctx, 256, 256, nullptr, 0, w1t, ctx_b1, bn1_g, bn1_b, bn1_m,
                  bn1_v, 1, tmpA, 128, N, stream);
  rungemm<128, 8>(tmpA, 128, 128, nullptr, 0, w2t, ctx_b2, bn2_g, bn2_b, bn2_m,
                  bn2_v, 1, fus, 192, N, stream);         // fus[:,0:128]
  rungemm<64, 4>(vis, 64, 64, nullptr, 0, wt1t, tgt_b1, nullptr, nullptr,
                 nullptr, nullptr, 1, z2, 64, N, stream);
  rungemm<64, 4>(z2, 64, 64, nullptr, 0, wt2t, tgt_b2, nullptr, nullptr,
                 nullptr, nullptr, 0, fus + 128, 192, N, stream); // fus[:,128:192]

  // ---- CSR build for both graphs ----------------------------------------
  const int eb = (E + 255) / 256;
  const int sb = (N + 1023) / 1024;
  k_count<<<eb, 256, 0, stream>>>(sp_ei + E, cnt_sp, E);
  k_count<<<eb, 256, 0, stream>>>(tr_ei + E, cnt_tr, E);
  k_scan1<<<sb, 1024, 0, stream>>>(cnt_sp, rp_sp, sums_sp, N);
  k_scan1<<<1, 1024, 0, stream>>>(sums_sp, sums_sp, nullptr, sb);
  k_scan_add<<<sb, 1024, 0, stream>>>(rp_sp, sums_sp, N);
  k_scan1<<<sb, 1024, 0, stream>>>(cnt_tr, rp_tr, sums_tr, N);
  k_scan1<<<1, 1024, 0, stream>>>(sums_tr, sums_tr, nullptr, sb);
  k_scan_add<<<sb, 1024, 0, stream>>>(rp_tr, sums_tr, N);
  k_dis<<<(N + 255) / 256, 256, 0, stream>>>(cnt_sp, dis_sp, N);
  k_dis<<<(N + 255) / 256, 256, 0, stream>>>(cnt_tr, dis_tr, N);
  k_fill<<<eb, 256, 0, stream>>>(sp_ei, sp_ei + E, rp_sp, cur_sp, col_sp, E);
  k_fill<<<eb, 256, 0, stream>>>(tr_ei, tr_ei + E, rp_tr, cur_tr, col_tr, E);

  // ---- GCN linear parts (bias/relu applied after aggregation) -----------
  rungemm<192, 4>(fus, 192, 192, nullptr, 0, wspt, nullptr, nullptr, nullptr,
                  nullptr, nullptr, 0, hsl, 64, N, stream);
  rungemm<192, 4>(fus, 192, 192, nullptr, 0, wtrt, nullptr, nullptr, nullptr,
                  nullptr, nullptr, 0, htl, 64, N, stream);

  // ---- aggregation (wave per destination node) --------------------------
  k_aggregate<<<(N + 7) / 8, 256, 0, stream>>>(rp_sp, cnt_sp, col_sp, dis_sp,
                                               hsl, spa_b, hcat, 128, 0, N);
  k_aggregate<<<(N + 7) / 8, 256, 0, stream>>>(rp_tr, cnt_tr, col_tr, dis_tr,
                                               htl, tra_b, hcat, 128, 64, N);

  // ---- fuse -> h (second output), then head -----------------------------
  rungemm<128, 4>(hcat, 128, 128, nullptr, 0, wfut, fuse_b, nullptr, nullptr,
                  nullptr, nullptr, 1, h_out, 64, N, stream);
  rungemm<256, 8>(h_out, 64, 64, fus, 192, wh1t, head_b1, bn3_g, bn3_b, bn3_m,
                  bn3_v, 1, tmpA, 128, N, stream);
  rungemm<128, 4>(tmpA, 128, 128, nullptr, 0, wh2t, head_b2, nullptr, nullptr,
                  nullptr, nullptr, 1, z2, 64, N, stream);
  rungemm<64, 4>(z2, 64, 64, nullptr, 0, wh3t, head_b3, nullptr, nullptr,
                 nullptr, nullptr, 0, y_out, 64, N, stream);
}
